// Qwen3NextGatedDeltaNet_49203145343531
// MI455X (gfx1250) — matrix operand order repaired and checked by reading, not compiled, not run on hardware
//
#include <hip/hip_runtime.h>
#include <hip/hip_bf16.h>
#include <math.h>

// ---------------------------------------------------------------------------
// Qwen3-Next Gated DeltaNet forward for MI455X (gfx1250, wave32, WMMA).
// GEMMs run on v_wmma_f32_16x16x32_bf16 with LDS tiles stored in fragment
// order (A row-major in k, B transposed) so every operand fetch is a wide
// ds_load_b128, plus a double-buffered global->LDS software pipeline.
// The recurrent delta-rule scan keeps per-thread state columns in VGPRs.
// ---------------------------------------------------------------------------

typedef __attribute__((ext_vector_type(16))) __bf16 v16bf;
typedef __attribute__((ext_vector_type(8)))  __bf16 v8bf;
typedef __attribute__((ext_vector_type(8)))  float  v8f;

#define L_SEQ   4096
#define HID     2048
#define HK_     16
#define HV_     32
#define DK_     128
#define DV_     128
#define NP1     12352      // QKVZ + BA columns of in_proj output
#define QKVZ_   12288
#define CONVD   8192       // 2*KEY_DIM + VAL_DIM
#define VALD    4096       // HV*DV
#define EPSF    1e-6f

// ---------------------------------------------------------------------------
// f32 -> bf16 conversion (for WMMA operands)
// ---------------------------------------------------------------------------
__global__ __launch_bounds__(256) void f32_to_bf16_kernel(const float* __restrict__ in,
                                                          __bf16* __restrict__ out,
                                                          size_t n) {
  size_t i = (size_t)blockIdx.x * blockDim.x + threadIdx.x;
  if (i < n) out[i] = (__bf16)in[i];
}

// ---------------------------------------------------------------------------
// Tiled bf16 WMMA GEMM: C[M,N] = A[M,K] * B[K,N], A/B bf16 row-major, C f32.
// 256 threads = 8 waves; block tile 128x128, K-step 32, double-buffered LDS.
// Wave grid 4(M) x 2(N): each wave owns a 32x64 tile = 2x4 WMMA 16x16 tiles.
//
// LDS layouts (so fragment fetches are contiguous 32B per lane):
//  As[buf][m][k]: A tile row-major, slot = k (no K permutation);
//     lane(half) reads
//     slots half*16 .. half*16+15 == K half*16..+15 (contiguous k order, same as Bt).
//  Bt[buf][n][k]: B tile transposed; lane(half) reads K half*16..+15 of col n.
// M,K are multiples of 128/32 at all call sites; N may be ragged.
// ---------------------------------------------------------------------------
#define TM 128
#define TN 128
#define TKK 32
#define TKP 40   // padded row length (bf16 elems) to stagger LDS banks

__global__ __launch_bounds__(256)
void gemm_bf16_wmma(const __bf16* __restrict__ A, const __bf16* __restrict__ B,
                    float* __restrict__ C, int M, int N, int Kd) {
  __shared__ __bf16 As[2][TM][TKP];   // 2 x 10 KB
  __shared__ __bf16 Bt[2][TN][TKP];   // 2 x 10 KB

  const int tid  = threadIdx.x;
  const int bm   = blockIdx.y * TM;
  const int bn   = blockIdx.x * TN;
  const int wave = tid >> 5;
  const int lane = tid & 31;
  const int wm   = (wave & 3) * 32;   // wave M offset inside block tile
  const int wn   = (wave >> 2) * 64;  // wave N offset inside block tile
  const int r    = lane & 15;
  const int half = lane >> 4;

  v8f acc[2][4] = {};

  // staging thread mapping
  const int arow = tid >> 1;           // 0..127 (A tile row)
  const int acol = (tid & 1) * 16;     // K sub-chunk 0 / 16
  const int albs = acol;               // LDS slot base for low 8 (0 or 16)
  const int bcol = tid & 127;          // 0..127 (B tile column)
  const int bkh  = tid >> 7;           // 0/1 -> K rows bkh*16 .. +15
  const bool fullN = (bn + TN <= N);

  const int ntile = Kd / TKK;

  // ---- helper lambdas: global loads into regs / reg stores into LDS ----
  auto load_a = [&](int k0, v8bf& lo, v8bf& hi) {
    const __bf16* src = A + (size_t)(bm + arow) * Kd + k0 + acol;
    lo = *(const v8bf*)(src);
    hi = *(const v8bf*)(src + 8);
  };
  auto load_b = [&](int k0, v16bf& bt) {
    if (fullN) {
      const __bf16* src = B + (size_t)(k0 + bkh * 16) * N + bn + bcol;
      #pragma unroll
      for (int i = 0; i < 16; ++i) bt[i] = src[(size_t)i * N];
    } else {
      const int col = bn + bcol;
      #pragma unroll
      for (int i = 0; i < 16; ++i)
        bt[i] = (col < N) ? B[(size_t)(k0 + bkh * 16 + i) * N + col] : (__bf16)0.f;
    }
  };
  auto store_tiles = [&](int buf, const v8bf& lo, const v8bf& hi, const v16bf& bt) {
    *(v8bf*)&As[buf][arow][albs]      = lo;   // K acol+0..7
    *(v8bf*)&As[buf][arow][albs + 8]  = hi;   // K acol+8..15
    *(v16bf*)&Bt[buf][bcol][bkh * 16] = bt;
  };

  // ---- prologue: stage tile 0 ----
  {
    v8bf lo, hi; v16bf bt;
    load_a(0, lo, hi);
    load_b(0, bt);
    store_tiles(0, lo, hi, bt);
  }
  __syncthreads();

  for (int kt = 0; kt < ntile; ++kt) {
    const int cur = kt & 1;
    const bool have_next = (kt + 1 < ntile);

    // issue next tile's global loads before the matrix work (pipelining)
    v8bf nlo = {}, nhi = {}; v16bf nbt = {};
    if (have_next) {
      load_a((kt + 1) * TKK, nlo, nhi);
      load_b((kt + 1) * TKK, nbt);
    }

    // ---- operand fragments: each is one contiguous 32B LDS read ----
    v16bf afrag[2];
    #pragma unroll
    for (int mt = 0; mt < 2; ++mt)
      afrag[mt] = *(const v16bf*)&As[cur][wm + mt * 16 + r][half * 16];
    v16bf bfrag[4];
    #pragma unroll
    for (int nt = 0; nt < 4; ++nt)
      bfrag[nt] = *(const v16bf*)&Bt[cur][wn + nt * 16 + r][half * 16];

    #pragma unroll
    for (int mt = 0; mt < 2; ++mt)
      #pragma unroll
      for (int nt = 0; nt < 4; ++nt)
        acc[mt][nt] = __builtin_amdgcn_wmma_f32_16x16x32_bf16(
            /*neg_a=*/false, afrag[mt], /*neg_b=*/false, bfrag[nt],
            /*c_mod=*/(short)0, acc[mt][nt],
            /*reuse_a=*/false, /*reuse_b=*/false);

    if (have_next) store_tiles(1 - cur, nlo, nhi, nbt);
    __syncthreads();
  }

  // ---- epilogue: C/D layout VGPR rr, lane (r,half): M = rr + 8*half, N = r ----
  #pragma unroll
  for (int mt = 0; mt < 2; ++mt) {
    #pragma unroll
    for (int nt = 0; nt < 4; ++nt) {
      const int col = bn + wn + nt * 16 + r;
      if (col < N) {
        #pragma unroll
        for (int rr = 0; rr < 8; ++rr) {
          const int row = bm + wm + mt * 16 + half * 8 + rr;
          C[(size_t)row * N + col] = acc[mt][nt][rr];
        }
      }
    }
  }
}

// ---------------------------------------------------------------------------
// Depthwise causal conv (K=4) + SiLU, with the qkvz->mixed_qkv column
// permutation applied on the fly (q|k|v flattened per-head layout).
// ---------------------------------------------------------------------------
__device__ __forceinline__ int mixed_to_proj_col(int c) {
  if (c < 2048) {                 // q: head h, dim d
    return (c >> 7) * 768 + (c & 127);
  } else if (c < 4096) {          // k
    int cc = c - 2048;
    return (cc >> 7) * 768 + 128 + (cc & 127);
  } else {                        // v: hv head -> (h = hv/2, sub = hv%2)
    int cc = c - 4096;
    int hv = cc >> 7, d = cc & 127;
    return (hv >> 1) * 768 + 256 + (hv & 1) * 128 + d;
  }
}

__global__ __launch_bounds__(256)
void conv_silu_kernel(const float* __restrict__ proj, const float* __restrict__ convw,
                      float* __restrict__ qkvc) {
  size_t idx = (size_t)blockIdx.x * 256 + threadIdx.x;
  int t = (int)(idx / CONVD);
  int c = (int)(idx % CONVD);
  const int pc = mixed_to_proj_col(c);
  const float4 w = *(const float4*)&convw[c * 4];   // 16B aligned
  float acc = 0.f;
  if (t >= 3) acc += w.x * proj[(size_t)(t - 3) * NP1 + pc];
  if (t >= 2) acc += w.y * proj[(size_t)(t - 2) * NP1 + pc];
  if (t >= 1) acc += w.z * proj[(size_t)(t - 1) * NP1 + pc];
  acc += w.w * proj[(size_t)t * NP1 + pc];
  qkvc[idx] = acc / (1.f + __expf(-acc));           // silu
}

// ---------------------------------------------------------------------------
// beta = sigmoid(b); g = -exp(A_log) * softplus(a + dt_bias)
// ---------------------------------------------------------------------------
__global__ __launch_bounds__(256)
void gate_coef_kernel(const float* __restrict__ proj, const float* __restrict__ A_log,
                      const float* __restrict__ dt_bias,
                      float* __restrict__ g, float* __restrict__ beta) {
  size_t idx = (size_t)blockIdx.x * 256 + threadIdx.x;
  if (idx >= (size_t)L_SEQ * HV_) return;
  int t = (int)(idx >> 5), hv = (int)(idx & 31);
  int h = hv >> 1, j = hv & 1;
  const float* row = proj + (size_t)t * NP1 + QKVZ_;
  float b = row[h * 4 + j];
  float a = row[h * 4 + 2 + j];
  beta[idx] = 1.f / (1.f + __expf(-b));
  float x = a + dt_bias[hv];
  float sp = (x > 20.f) ? x : log1pf(__expf(x));
  g[idx] = -__expf(A_log[hv]) * sp;
}

// ---------------------------------------------------------------------------
// Recurrent delta-rule scan. One block per v-head (32 blocks), 128 threads.
// Thread d owns state column S[:,d] in 128 VGPRs (fully unrolled loops).
// Per step: l2norm(q,k) via shfl+LDS reduction, then
//   S = S*exp(g); delta = (v - k.S)*beta; S += k (x) delta; o = q.S
// kn/qn broadcasts are read as float4 to minimize ds ops on the serial path.
// ---------------------------------------------------------------------------
__global__ __launch_bounds__(128)
void gdn_scan_kernel(const float* __restrict__ qkvc, const float* __restrict__ g,
                     const float* __restrict__ beta, float* __restrict__ core) {
  const int hv = blockIdx.x;
  const int hk = hv >> 1;      // q,k are repeated G=2x over v-heads
  const int d  = threadIdx.x;  // 0..127  (DV column owned by this thread)
  const int wid = d >> 5, ln = d & 31;

  __shared__ float kn[DK_];
  __shared__ float qn[DK_];
  __shared__ float redq[4], redk[4];

  float S[DK_];
  #pragma unroll
  for (int i = 0; i < DK_; ++i) S[i] = 0.f;

  const int qoff = hk * 128;
  const int koff = 2048 + hk * 128;
  const int voff = 4096 + hv * 128;
  const float qscale = 0.08838834764831845f;  // DK^-0.5

  for (int t = 0; t < L_SEQ; ++t) {
    const float* row = qkvc + (size_t)t * CONVD;
    const float qd = row[qoff + d];
    const float kd = row[koff + d];
    const float vd = row[voff + d];

    // block-wide sum of q^2 and k^2 (wave32 shfl tree + 4-wave LDS combine)
    float sq = qd * qd, sk = kd * kd;
    #pragma unroll
    for (int off = 16; off > 0; off >>= 1) {
      sq += __shfl_xor(sq, off, 32);
      sk += __shfl_xor(sk, off, 32);
    }
    if (ln == 0) { redq[wid] = sq; redk[wid] = sk; }
    __syncthreads();
    sq = redq[0] + redq[1] + redq[2] + redq[3];
    sk = redk[0] + redk[1] + redk[2] + redk[3];

    qn[d] = qd * rsqrtf(sq + EPSF) * qscale;
    kn[d] = kd * rsqrtf(sk + EPSF);
    __syncthreads();

    const float gt    = g[(size_t)t * HV_ + hv];
    const float bt    = beta[(size_t)t * HV_ + hv];
    const float decay = __expf(gt);

    // pass 1: ksum = k . (decay*S_old)[:,d]   (float4 LDS broadcasts)
    float ksum = 0.f;
    #pragma unroll
    for (int j = 0; j < DK_ / 4; ++j) {
      const float4 kv = *(const float4*)&kn[4 * j];
      ksum = fmaf(kv.x, S[4 * j + 0], ksum);
      ksum = fmaf(kv.y, S[4 * j + 1], ksum);
      ksum = fmaf(kv.z, S[4 * j + 2], ksum);
      ksum = fmaf(kv.w, S[4 * j + 3], ksum);
    }
    ksum *= decay;
    const float delta = (vd - ksum) * bt;

    // pass 2: S = decay*S + k (x) delta ; o = q . S_new
    float osum = 0.f;
    #pragma unroll
    for (int j = 0; j < DK_ / 4; ++j) {
      const float4 kv = *(const float4*)&kn[4 * j];
      const float4 qv = *(const float4*)&qn[4 * j];
      S[4 * j + 0] = fmaf(S[4 * j + 0], decay, kv.x * delta);
      S[4 * j + 1] = fmaf(S[4 * j + 1], decay, kv.y * delta);
      S[4 * j + 2] = fmaf(S[4 * j + 2], decay, kv.z * delta);
      S[4 * j + 3] = fmaf(S[4 * j + 3], decay, kv.w * delta);
      osum = fmaf(qv.x, S[4 * j + 0], osum);
      osum = fmaf(qv.y, S[4 * j + 1], osum);
      osum = fmaf(qv.z, S[4 * j + 2], osum);
      osum = fmaf(qv.w, S[4 * j + 3], osum);
    }
    core[((size_t)t * HV_ + hv) * DV_ + d] = osum;
    __syncthreads();  // protect kn/qn/red for next step
  }
}

// ---------------------------------------------------------------------------
// RMSNorm over DV, * norm_w, * silu(z); emit bf16 activations for out_proj.
// Grid: (L, HV), 128 threads.
// ---------------------------------------------------------------------------
__global__ __launch_bounds__(128)
void norm_gate_kernel(const float* __restrict__ core, const float* __restrict__ proj,
                      const float* __restrict__ norm_w, __bf16* __restrict__ act) {
  const int t = blockIdx.x, hv = blockIdx.y, d = threadIdx.x;
  const int wid = d >> 5, ln = d & 31;
  __shared__ float red[4];

  const float x = core[((size_t)t * HV_ + hv) * DV_ + d];
  float s = x * x;
  #pragma unroll
  for (int off = 16; off > 0; off >>= 1) s += __shfl_xor(s, off, 32);
  if (ln == 0) red[wid] = s;
  __syncthreads();
  s = red[0] + red[1] + red[2] + red[3];
  const float inv = rsqrtf(s * (1.f / DV_) + EPSF);

  const int h = hv >> 1, sub = hv & 1;
  const float z = proj[(size_t)t * NP1 + h * 768 + 512 + sub * 128 + d];
  const float y = x * inv * norm_w[d] * (z / (1.f + __expf(-z)));
  act[(size_t)t * VALD + hv * 128 + d] = (__bf16)y;
}

// ---------------------------------------------------------------------------
// Host-side launch.
// Workspace layout (~520 MB assumed available in d_ws):
//   hs_bf16 | w1_bf16 | w2_bf16 | proj f32 | qkvc f32 | g | beta | core | act
// ---------------------------------------------------------------------------
static inline size_t align256(size_t x) { return (x + 255) & ~(size_t)255; }

extern "C" void kernel_launch(void* const* d_in, const int* in_sizes, int n_in,
                              void* d_out, int out_size, void* d_ws, size_t ws_size,
                              hipStream_t stream) {
  const float* hs      = (const float*)d_in[0];
  const float* w1      = (const float*)d_in[1];
  const float* convw   = (const float*)d_in[2];
  const float* A_log   = (const float*)d_in[3];
  const float* dt_bias = (const float*)d_in[4];
  const float* norm_w  = (const float*)d_in[5];
  const float* w2      = (const float*)d_in[6];
  float* out = (float*)d_out;

  char* ws = (char*)d_ws;
  size_t off = 0;
  __bf16* hsb = (__bf16*)(ws + off); off = align256(off + (size_t)L_SEQ * HID * 2);
  __bf16* w1b = (__bf16*)(ws + off); off = align256(off + (size_t)HID * NP1 * 2);
  __bf16* w2b = (__bf16*)(ws + off); off = align256(off + (size_t)VALD * HID * 2);
  float* proj = (float*)(ws + off);  off = align256(off + (size_t)L_SEQ * NP1 * 4);
  float* qkvc = (float*)(ws + off);  off = align256(off + (size_t)L_SEQ * CONVD * 4);
  float* gbuf = (float*)(ws + off);  off = align256(off + (size_t)L_SEQ * HV_ * 4);
  float* bbuf = (float*)(ws + off);  off = align256(off + (size_t)L_SEQ * HV_ * 4);
  float* core = (float*)(ws + off);  off = align256(off + (size_t)L_SEQ * HV_ * DV_ * 4);
  __bf16* act = (__bf16*)(ws + off); off = align256(off + (size_t)L_SEQ * VALD * 2);
  (void)ws_size; (void)in_sizes; (void)n_in; (void)out_size;

  // 1. convert GEMM operands to bf16
  {
    size_t n;
    n = (size_t)L_SEQ * HID;
    f32_to_bf16_kernel<<<(n + 255) / 256, 256, 0, stream>>>(hs, hsb, n);
    n = (size_t)HID * NP1;
    f32_to_bf16_kernel<<<(n + 255) / 256, 256, 0, stream>>>(w1, w1b, n);
    n = (size_t)VALD * HID;
    f32_to_bf16_kernel<<<(n + 255) / 256, 256, 0, stream>>>(w2, w2b, n);
  }

  // 2. in_proj GEMM: [L,H] x [H,NP1] -> proj
  {
    dim3 grid((NP1 + TN - 1) / TN, L_SEQ / TM);
    gemm_bf16_wmma<<<grid, 256, 0, stream>>>(hsb, w1b, proj, L_SEQ, NP1, HID);
  }

  // 3. conv + silu (with qkvz -> mixed_qkv permutation fused)
  conv_silu_kernel<<<((size_t)L_SEQ * CONVD) / 256, 256, 0, stream>>>(proj, convw, qkvc);

  // 4. beta / g coefficients
  gate_coef_kernel<<<((size_t)L_SEQ * HV_ + 255) / 256, 256, 0, stream>>>(
      proj, A_log, dt_bias, gbuf, bbuf);

  // 5. recurrent delta-rule scan (one block per v-head)
  gdn_scan_kernel<<<HV_, 128, 0, stream>>>(qkvc, gbuf, bbuf, core);

  // 6. RMSNorm + gate -> bf16 activations
  {
    dim3 grid(L_SEQ, HV_);
    norm_gate_kernel<<<grid, 128, 0, stream>>>(core, proj, norm_w, act);
  }

  // 7. out_proj GEMM: [L,VALD] x [VALD,H] -> out (f32)
  {
    dim3 grid(HID / TN, L_SEQ / TM);
    gemm_bf16_wmma<<<grid, 256, 0, stream>>>(act, w2b, out, L_SEQ, HID, VALD);
  }
}
